// g2g_JTMPN_57492432224748
// MI455X (gfx1250) — matrix operand order repaired and checked by reading, not compiled, not run on hardware
//
#include <hip/hip_runtime.h>
#include <hip/hip_bf16.h>
#include <stdint.h>

// ---------------------------------------------------------------------------
// Sizes (fixed by the reference problem)
// ---------------------------------------------------------------------------
#define N_NODES 100000
#define N_EDGES 300000   // divisible by 16
#define N_MSG   10000    // divisible by 16
#define DN 256
#define DE 64
#define DM 256
#define DX 256

typedef __attribute__((ext_vector_type(16))) __bf16 bf16x16;
typedef __attribute__((ext_vector_type(8)))  float  floatx8;

static __device__ __forceinline__ floatx8 wmma_bf16(bf16x16 a, bf16x16 b, floatx8 c) {
  // D = A(16x32 bf16) x B(32x16 bf16) + C(16x16 f32)
  return __builtin_amdgcn_wmma_f32_16x16x32_bf16(false, a, false, b, (short)0, c,
                                                 false, false);
}

// A-matrix (16-bit, 16x32) lane layout: lane L holds row m=L%16; VGPR0..3 hold
// K = half*8 + [0..8), VGPR4..7 hold K = 16 + half*8 + [0..8)  (half = L/16).
static __device__ __forceinline__ bf16x16 load_a_f32(const float* __restrict__ row,
                                                     int k0, int half) {
  const float* p0 = row + k0 + half * 8;
  bf16x16 a;
#pragma unroll
  for (int t = 0; t < 8; ++t) a[t] = (__bf16)p0[t];
#pragma unroll
  for (int t = 0; t < 8; ++t) a[8 + t] = (__bf16)p0[16 + t];
  return a;
}

static __device__ __forceinline__ bf16x16 load_a_lds(const float* __restrict__ s,
                                                     int m, int k0, int half) {
  const float* p0 = s + m * 260 + k0 + half * 8;  // 260-float row stride (bank pad)
  bf16x16 a;
#pragma unroll
  for (int t = 0; t < 8; ++t) a[t] = (__bf16)p0[t];
#pragma unroll
  for (int t = 0; t < 8; ++t) a[8 + t] = (__bf16)p0[16 + t];
  return a;
}

// B tiles prepacked so each lane reads 8 contiguous dwords.
static __device__ __forceinline__ bf16x16 load_b_packed(const uint32_t* __restrict__ wp,
                                                        int tileIdx, int lane) {
  const uint32_t* p = wp + (((size_t)tileIdx * 32 + lane) * 8);
  union { uint32_t u[8]; bf16x16 v; } x;
#pragma unroll
  for (int j = 0; j < 8; ++j) x.u[j] = p[j];
  return x.v;
}

// ---------------------------------------------------------------------------
// Weight prepack: fp32 (K x Nw, row major) -> bf16 WMMA-B lane layout.
// B (32x16) layout: half=lane/16; VGPR0..3 hold K = half*8 + [0..8), VGPR4..7 hold
// K = 16 + half*8 + [0..8) (k pairs lo/hi, col = lane%16). Tile id = kt*(Nw/16)+nt.
// ---------------------------------------------------------------------------
__global__ void pack_weight_kernel(const float* __restrict__ W,
                                   uint32_t* __restrict__ out, int K, int Nw) {
  int gid = blockIdx.x * blockDim.x + threadIdx.x;
  int NT = Nw >> 4;
  int total = (K >> 5) * NT * 32;
  if (gid >= total) return;
  int lane = gid & 31;
  int tid  = gid >> 5;
  int nt = tid % NT;
  int kt = tid / NT;
  int kg = lane >> 4;
  int n  = lane & 15;
  uint32_t* dst = out + ((size_t)tid * 32 + lane) * 8;
#pragma unroll
  for (int j = 0; j < 8; ++j) {
    int k = kt * 32 + kg * 8 + (j >> 2) * 16 + 2 * (j & 3);
    union { __bf16 h[2]; uint32_t u; } p;
    p.h[0] = (__bf16)W[(size_t)k * Nw + nt * 16 + n];
    p.h[1] = (__bf16)W[(size_t)(k + 1) * Nw + nt * 16 + n];
    dst[j] = p.u;
  }
}

// ---------------------------------------------------------------------------
// Elementwise helpers
// ---------------------------------------------------------------------------
__global__ void zero_f32_kernel(float* __restrict__ p, long n4) {
  long i = (long)blockIdx.x * blockDim.x + threadIdx.x;
  if (i < n4) *(float4*)(p + i * 4) = make_float4(0.f, 0.f, 0.f, 0.f);
}

__global__ void relu_copy_kernel(const float* __restrict__ src,
                                 float* __restrict__ dst, long n4) {
  long i = (long)blockIdx.x * blockDim.x + threadIdx.x;
  if (i < n4) {
    float4 v = *(const float4*)(src + i * 4);
    v.x = fmaxf(v.x, 0.f); v.y = fmaxf(v.y, 0.f);
    v.z = fmaxf(v.z, 0.f); v.w = fmaxf(v.w, 0.f);
    *(float4*)(dst + i * 4) = v;
  }
}

// segment_sum(msg, edge_dst) via f32 atomics (agg is L2-resident: 102MB < 192MB L2)
__global__ void scatter_add_kernel(const float* __restrict__ msg,
                                   const int* __restrict__ edge_dst,
                                   float* __restrict__ agg) {
  int gid = blockIdx.x * blockDim.x + threadIdx.x;   // 64 threads per edge
  int e  = gid >> 6;
  int c  = (gid & 63) << 2;
  if (e >= N_EDGES) return;
  int d = edge_dst[e];
  float4 v = *(const float4*)(msg + (size_t)e * DM + c);
  float* dst = agg + (size_t)d * DM + c;
  unsafeAtomicAdd(dst + 0, v.x);
  unsafeAtomicAdd(dst + 1, v.y);
  unsafeAtomicAdd(dst + 2, v.z);
  unsafeAtomicAdd(dst + 3, v.w);
}

// ---------------------------------------------------------------------------
// base = f_node[edge_src] @ W1 + f_edge @ W2 + b1
// block = 128 threads (4 waves); each wave: 16 rows x 64 cols strip.
// ---------------------------------------------------------------------------
__global__ void __launch_bounds__(128) base_kernel(
    const float* __restrict__ f_node, const float* __restrict__ f_edge,
    const int* __restrict__ edge_src, const float* __restrict__ b1,
    const uint32_t* __restrict__ W1p, const uint32_t* __restrict__ W2p,
    float* __restrict__ base) {
  const int rt   = blockIdx.x;
  const int lane = threadIdx.x & 31;
  const int w    = threadIdx.x >> 5;
  const int m    = lane & 15;
  const int half = lane >> 4;
  const int e    = rt * 16 + m;
  const int nb   = w * 64;
  const float* arow = f_node + (size_t)edge_src[e] * DN;
  const float* erow = f_edge + (size_t)e * DE;

  floatx8 acc[4];
#pragma unroll
  for (int j = 0; j < 4; ++j) {
    float bv = b1[nb + j * 16 + m];
#pragma unroll
    for (int i = 0; i < 8; ++i) acc[j][i] = bv;
  }
#pragma unroll
  for (int kc = 0; kc < DN / 32; ++kc) {           // f_node @ W1, K=256
    bf16x16 a = load_a_f32(arow, kc * 32, half);
#pragma unroll
    for (int j = 0; j < 4; ++j) {
      bf16x16 b = load_b_packed(W1p, kc * (DM / 16) + w * 4 + j, lane);
      acc[j] = wmma_bf16(a, b, acc[j]);
    }
  }
#pragma unroll
  for (int kc = 0; kc < DE / 32; ++kc) {           // f_edge @ W2, K=64
    bf16x16 a = load_a_f32(erow, kc * 32, half);
#pragma unroll
    for (int j = 0; j < 4; ++j) {
      bf16x16 b = load_b_packed(W2p, kc * (DM / 16) + w * 4 + j, lane);
      acc[j] = wmma_bf16(a, b, acc[j]);
    }
  }
#pragma unroll
  for (int j = 0; j < 4; ++j)
#pragma unroll
    for (int i = 0; i < 8; ++i)
      base[(size_t)(rt * 16 + i + 8 * half) * DM + nb + j * 16 + m] = acc[j][i];
}

// ---------------------------------------------------------------------------
// base[tree_tgt] += tree_msg @ W3   (alpha@W3 folded into base; targets unique)
// ---------------------------------------------------------------------------
__global__ void __launch_bounds__(128) alpha_fold_kernel(
    const float* __restrict__ tree_msg, const int* __restrict__ tree_tgt,
    const uint32_t* __restrict__ W3p, float* __restrict__ base) {
  const int rt   = blockIdx.x;
  const int lane = threadIdx.x & 31;
  const int w    = threadIdx.x >> 5;
  const int m    = lane & 15;
  const int half = lane >> 4;
  const int nb   = w * 64;
  const float* arow = tree_msg + (size_t)(rt * 16 + m) * DM;

  floatx8 acc[4] = {};
#pragma unroll
  for (int kc = 0; kc < DM / 32; ++kc) {
    bf16x16 a = load_a_f32(arow, kc * 32, half);
#pragma unroll
    for (int j = 0; j < 4; ++j) {
      bf16x16 b = load_b_packed(W3p, kc * (DM / 16) + w * 4 + j, lane);
      acc[j] = wmma_bf16(a, b, acc[j]);
    }
  }
  int tg[8];
#pragma unroll
  for (int i = 0; i < 8; ++i) tg[i] = tree_tgt[rt * 16 + i + 8 * half];
#pragma unroll
  for (int j = 0; j < 4; ++j)
#pragma unroll
    for (int i = 0; i < 8; ++i)
      base[(size_t)tg[i] * DM + nb + j * 16 + m] += acc[j][i];
}

// ---------------------------------------------------------------------------
// msg = relu(base + (agg[edge_src] - msg[e^1]) @ W3), IN PLACE.
// Safe: rev(e)=e^1 is inside this block's 16-row tile; block reads all of its
// msg inputs into LDS before any write; agg is from a prior kernel.
// ---------------------------------------------------------------------------
__global__ void __launch_bounds__(128) msg_update_kernel(
    float* __restrict__ msg, const float* __restrict__ agg,
    const float* __restrict__ base, const int* __restrict__ edge_src,
    const uint32_t* __restrict__ W3p) {
  __shared__ float sm[16 * 260];
  const int rt  = blockIdx.x;
  const int tid = threadIdx.x;
  {   // phase 1: sum_msg rows -> LDS (16 x 256, padded stride 260)
    const int r  = tid >> 3;
    const int c0 = (tid & 7) * 32;
    const int e  = rt * 16 + r;
    const float* ar = agg + (size_t)edge_src[e] * DM;
    const float* mr = msg + (size_t)(e ^ 1) * DM;
    float* srow = sm + r * 260;
#pragma unroll
    for (int c = 0; c < 32; c += 4) {
      float4 av = *(const float4*)(ar + c0 + c);
      float4 mv = *(const float4*)(mr + c0 + c);
      *(float4*)(srow + c0 + c) =
          make_float4(av.x - mv.x, av.y - mv.y, av.z - mv.z, av.w - mv.w);
    }
  }
  __syncthreads();
  const int lane = tid & 31;
  const int w    = tid >> 5;
  const int m    = lane & 15;
  const int half = lane >> 4;
  const int nb   = w * 64;

  floatx8 acc[4];
#pragma unroll
  for (int j = 0; j < 4; ++j)
#pragma unroll
    for (int i = 0; i < 8; ++i)
      acc[j][i] = base[(size_t)(rt * 16 + i + 8 * half) * DM + nb + j * 16 + m];

#pragma unroll
  for (int kc = 0; kc < DM / 32; ++kc) {
    bf16x16 a = load_a_lds(sm, m, kc * 32, half);
#pragma unroll
    for (int j = 0; j < 4; ++j) {
      bf16x16 b = load_b_packed(W3p, kc * (DM / 16) + w * 4 + j, lane);
      acc[j] = wmma_bf16(a, b, acc[j]);
    }
  }
#pragma unroll
  for (int j = 0; j < 4; ++j)
#pragma unroll
    for (int i = 0; i < 8; ++i)
      msg[(size_t)(rt * 16 + i + 8 * half) * DM + nb + j * 16 + m] =
          fmaxf(acc[j][i], 0.0f);
}

// ---------------------------------------------------------------------------
// x = relu(f_node @ W4 + node_sum @ W5 + b2); one wave per 16x64 strip.
// ---------------------------------------------------------------------------
__global__ void __launch_bounds__(256) readout_kernel(
    const float* __restrict__ f_node, const float* __restrict__ node_sum,
    const uint32_t* __restrict__ W4p, const uint32_t* __restrict__ W5p,
    const float* __restrict__ b2, float* __restrict__ x) {
  const int gw   = blockIdx.x * 8 + (threadIdx.x >> 5);
  const int rt   = gw >> 2;
  const int w    = gw & 3;
  const int lane = threadIdx.x & 31;
  const int m    = lane & 15;
  const int half = lane >> 4;
  const int nb   = w * 64;
  const float* nrow = f_node   + (size_t)(rt * 16 + m) * DN;
  const float* srow = node_sum + (size_t)(rt * 16 + m) * DM;

  floatx8 acc[4];
#pragma unroll
  for (int j = 0; j < 4; ++j) {
    float bv = b2[nb + j * 16 + m];
#pragma unroll
    for (int i = 0; i < 8; ++i) acc[j][i] = bv;
  }
#pragma unroll
  for (int kc = 0; kc < DN / 32; ++kc) {
    bf16x16 a = load_a_f32(nrow, kc * 32, half);
#pragma unroll
    for (int j = 0; j < 4; ++j) {
      bf16x16 b = load_b_packed(W4p, kc * (DX / 16) + w * 4 + j, lane);
      acc[j] = wmma_bf16(a, b, acc[j]);
    }
  }
#pragma unroll
  for (int kc = 0; kc < DM / 32; ++kc) {
    bf16x16 a = load_a_f32(srow, kc * 32, half);
#pragma unroll
    for (int j = 0; j < 4; ++j) {
      bf16x16 b = load_b_packed(W5p, kc * (DX / 16) + w * 4 + j, lane);
      acc[j] = wmma_bf16(a, b, acc[j]);
    }
  }
#pragma unroll
  for (int j = 0; j < 4; ++j)
#pragma unroll
    for (int i = 0; i < 8; ++i)
      x[(size_t)(rt * 16 + i + 8 * half) * DX + nb + j * 16 + m] =
          fmaxf(acc[j][i], 0.0f);
}

// ---------------------------------------------------------------------------
// Host driver
// ---------------------------------------------------------------------------
extern "C" void kernel_launch(void* const* d_in, const int* in_sizes, int n_in,
                              void* d_out, int out_size, void* d_ws, size_t ws_size,
                              hipStream_t stream) {
  const float* f_node   = (const float*)d_in[0];
  const float* f_edge   = (const float*)d_in[1];
  const float* tree_msg = (const float*)d_in[2];
  const float* W1       = (const float*)d_in[3];
  const float* W2       = (const float*)d_in[4];
  const float* W3       = (const float*)d_in[5];
  const float* b1       = (const float*)d_in[6];
  const float* W4       = (const float*)d_in[7];
  const float* W5       = (const float*)d_in[8];
  const float* b2       = (const float*)d_in[9];
  const int*   edge_src = (const int*)d_in[10];
  const int*   edge_dst = (const int*)d_in[11];
  const int*   tree_tgt = (const int*)d_in[12];
  float* out = (float*)d_out;

  char* ws = (char*)d_ws;
  size_t off = 0;
  auto wsAlloc = [&](size_t bytes) -> void* {
    void* p = ws + off;
    off = (off + bytes + 255) & ~(size_t)255;
    return p;
  };
  float*    base = (float*)wsAlloc((size_t)N_EDGES * DM * sizeof(float)); // 307.2 MB
  float*    msg  = (float*)wsAlloc((size_t)N_EDGES * DM * sizeof(float)); // 307.2 MB
  float*    agg  = (float*)wsAlloc((size_t)N_NODES * DM * sizeof(float)); // 102.4 MB
  uint32_t* W1p  = (uint32_t*)wsAlloc((size_t)DN * DM * 2);
  uint32_t* W2p  = (uint32_t*)wsAlloc((size_t)DE * DM * 2);
  uint32_t* W3p  = (uint32_t*)wsAlloc((size_t)DM * DM * 2);
  uint32_t* W4p  = (uint32_t*)wsAlloc((size_t)DN * DX * 2);
  uint32_t* W5p  = (uint32_t*)wsAlloc((size_t)DM * DX * 2);

  auto pack = [&](const float* W, uint32_t* Wp, int K, int Nw) {
    int threads = (K / 32) * (Nw / 16) * 32;
    pack_weight_kernel<<<(threads + 255) / 256, 256, 0, stream>>>(W, Wp, K, Nw);
  };
  pack(W1, W1p, DN, DM);
  pack(W2, W2p, DE, DM);
  pack(W3, W3p, DM, DM);
  pack(W4, W4p, DN, DX);
  pack(W5, W5p, DM, DX);

  const long edge_n4 = (long)N_EDGES * DM / 4;   // 19.2M float4
  const long node_n4 = (long)N_NODES * DM / 4;   //  6.4M float4
  const int  scat_blocks = (N_EDGES * 64) / 256; // 64 threads/edge

  // base = gatherGEMM + GEMM + b1 ; then fold alpha@W3 into it
  base_kernel<<<N_EDGES / 16, 128, 0, stream>>>(f_node, f_edge, edge_src, b1,
                                                W1p, W2p, base);
  alpha_fold_kernel<<<N_MSG / 16, 128, 0, stream>>>(tree_msg, tree_tgt, W3p, base);

  // iteration 1 (msg was 0): msg = relu(base)
  relu_copy_kernel<<<(int)((edge_n4 + 255) / 256), 256, 0, stream>>>(base, msg, edge_n4);

  // iterations 2..3
  for (int it = 0; it < 2; ++it) {
    zero_f32_kernel<<<(int)((node_n4 + 255) / 256), 256, 0, stream>>>(agg, node_n4);
    scatter_add_kernel<<<scat_blocks, 256, 0, stream>>>(msg, edge_dst, agg);
    msg_update_kernel<<<N_EDGES / 16, 128, 0, stream>>>(msg, agg, base, edge_src, W3p);
  }

  // node_sum = segment_sum(msg, edge_dst); readout
  zero_f32_kernel<<<(int)((node_n4 + 255) / 256), 256, 0, stream>>>(agg, node_n4);
  scatter_add_kernel<<<scat_blocks, 256, 0, stream>>>(msg, edge_dst, agg);
  readout_kernel<<<(N_NODES / 16) * 4 / 8, 256, 0, stream>>>(f_node, agg, W4p, W5p,
                                                             b2, out);
}
